// GPT3Attention_67860483277115
// MI455X (gfx1250) — matrix operand order repaired and checked by reading, not compiled, not run on hardware
//
#include <hip/hip_runtime.h>

// ---- problem constants (match reference) -----------------------------------
#define Bq 2
#define Sq 4096
#define Dq 512
#define Hq 8
#define Fq 2048
#define DHq 64
#define EPSq 1e-3f

typedef __attribute__((ext_vector_type(16))) __bf16 bf16x16;
typedef __attribute__((ext_vector_type(8)))  __bf16 bf16x8;
typedef __attribute__((ext_vector_type(4)))  __bf16 bf16x4;
typedef __attribute__((ext_vector_type(8)))  float  f32x8;

// ---------------------------------------------------------------------------
// Fragment loaders: each fragment = two 16-byte LDS vector loads (ds_load_b128)
// A 16-bit frag (ISA 7.12.2): elements 0..7 at K=8h+e, 8..15 at K=16+8h+e-8
// B 16-bit frag:              same as A: 0..7 at K=8h+e, 8..15 at K=16+8h+e-8
// `win` points at the start of the 32-wide K window for this lane's row/col.
// ---------------------------------------------------------------------------
__device__ __forceinline__ bf16x16 load_a_frag(const __bf16* win, int half) {
  union { bf16x16 v; bf16x8 h[2]; } u;
  u.h[0] = *(const bf16x8*)(win + 8 * half);
  u.h[1] = *(const bf16x8*)(win + 16 + 8 * half);
  return u.v;
}
__device__ __forceinline__ bf16x16 load_b_frag(const __bf16* win, int half) {
  union { bf16x16 v; bf16x8 h[2]; } u;
  u.h[0] = *(const bf16x8*)(win + 8 * half);
  u.h[1] = *(const bf16x8*)(win + 16 + 8 * half);
  return u.v;
}
__device__ __forceinline__ void store_bf4(__bf16* dst, float a, float b,
                                          float c, float d) {
  bf16x4 t;
  t[0] = (__bf16)a; t[1] = (__bf16)b; t[2] = (__bf16)c; t[3] = (__bf16)d;
  *(bf16x4*)dst = t;   // ds_store_b64
}

// ============================================================================
// Generic GEMM: C = act( A[M,K] @ W[K,N] + bias (+ residual) )
// block tile 128x64, K-step 32, 8 waves; wave w owns rows [w*16, w*16+16).
// Double-buffered LDS (one barrier per K-step) + register-staged global
// prefetch (next tile's vmem overlaps this tile's WMMAs). W tile stored
// transposed (K-innermost) so B-fragments are two contiguous b128 loads.
// ============================================================================
#define GBM 128
#define GBN 64
#define GBK 32
#define APITCH (GBK + 8)   // 40 bf16 = 80 B (16B-aligned rows)
#define WPITCH (GBK + 8)

__global__ __launch_bounds__(256) void gemm_bf16_wmma(
    const float* __restrict__ A, const float* __restrict__ Wm,
    const float* __restrict__ bias, const float* __restrict__ residual,
    float* __restrict__ Cc, int M, int N, int K, int relu)
{
  __shared__ __bf16 As[2][GBM][APITCH];     // [buf][row][k]
  __shared__ __bf16 WsT[2][GBN][WPITCH];    // [buf][col][k]  (transposed)
  (void)M;

  const int tid    = threadIdx.x;
  const int wv     = tid >> 5;
  const int lane   = tid & 31;
  const int lanelo = lane & 15;
  const int half   = lane >> 4;
  const int m0     = blockIdx.y * GBM;
  const int n0     = blockIdx.x * GBN;

  f32x8 acc[4];
  #pragma unroll
  for (int j = 0; j < 4; ++j)
    #pragma unroll
    for (int e = 0; e < 8; ++e) acc[j][e] = 0.f;

  float4 aR[4];      // A staging: 4 x float4 per thread (128x32 tile)
  float  wR[2][4];   // W staging: 2 x 4 strided floats per thread (32x64 tile)

  auto load_tiles = [&](int k0c) {
    #pragma unroll
    for (int it = 0; it < 4; ++it) {
      int idx = tid + it * 256;            // 0..1023
      int i = idx >> 3, g = idx & 7;       // row, k-group of 4
      aR[it] = *(const float4*)&A[(size_t)(m0 + i) * K + k0c + 4 * g];
    }
    #pragma unroll
    for (int it = 0; it < 2; ++it) {
      int idx = tid + it * 256;            // 0..511
      int n = idx & 63, g = idx >> 6;      // col, k-group of 4
      #pragma unroll
      for (int kk = 0; kk < 4; ++kk)
        wR[it][kk] = Wm[(size_t)(k0c + 4 * g + kk) * N + n0 + n];
    }
  };
  auto store_tiles = [&](int p) {
    #pragma unroll
    for (int it = 0; it < 4; ++it) {
      int idx = tid + it * 256;
      int i = idx >> 3, g = idx & 7;
      store_bf4(&As[p][i][4 * g], aR[it].x, aR[it].y, aR[it].z, aR[it].w);
    }
    #pragma unroll
    for (int it = 0; it < 2; ++it) {
      int idx = tid + it * 256;
      int n = idx & 63, g = idx >> 6;
      store_bf4(&WsT[p][n][4 * g], wR[it][0], wR[it][1], wR[it][2], wR[it][3]);
    }
  };

  load_tiles(0);
  store_tiles(0);
  __syncthreads();

  int p = 0;
  for (int k0 = 0; k0 < K; k0 += GBK) {
    const bool more = (k0 + GBK) < K;
    if (more) load_tiles(k0 + GBK);        // vmem overlaps WMMAs below

    bf16x16 af = load_a_frag(&As[p][wv * 16 + lanelo][0], half);
    bf16x16 bf4v[4];
    #pragma unroll
    for (int j = 0; j < 4; ++j)            // all 8 b128 loads first ...
      bf4v[j] = load_b_frag(&WsT[p][j * 16 + lanelo][0], half);
    #pragma unroll
    for (int j = 0; j < 4; ++j)            // ... then 4 back-to-back WMMAs
      acc[j] = __builtin_amdgcn_wmma_f32_16x16x32_bf16(
          false, af, false, bf4v[j], (short)0, acc[j], false, false);

    if (more) store_tiles(p ^ 1);          // other buffer: no fence needed
    __syncthreads();
    p ^= 1;
  }

  #pragma unroll
  for (int j = 0; j < 4; ++j) {
    int col = n0 + j * 16 + lanelo;
    #pragma unroll
    for (int r = 0; r < 8; ++r) {
      int row = m0 + wv * 16 + r + 8 * half;
      float v = acc[j][r] + bias[col];
      if (residual) v += residual[(size_t)row * N + col];
      if (relu) v = fmaxf(v, 0.f);
      Cc[(size_t)row * N + col] = v;
    }
  }
}

// ============================================================================
// Flash attention (causal), per (b,h): 128 queries/block, 64-key tiles.
// Qs/Ks K-innermost (dh); Vs stored transposed (key-innermost) so the P@V
// B-fragments are contiguous. Never materializes the [S,S] score tensor.
// ============================================================================
#define FQB 128
#define FKB 64
#define QPITCH (DHq + 8)   // 72 bf16 = 144 B (16B-aligned rows)
#define VPITCH (FKB + 8)

__global__ __launch_bounds__(256) void flash_attn_wmma(
    const float* __restrict__ Qg, const float* __restrict__ Kg,
    const float* __restrict__ Vg, float* __restrict__ Ctx)
{
  __shared__ __bf16 Qs[FQB][QPITCH];      // [query][dh]
  __shared__ __bf16 Ks[FKB][QPITCH];      // [key][dh]
  __shared__ __bf16 VsT[DHq][VPITCH];     // [dh][key]  (transposed)
  __shared__ __bf16 Ps[8][16][VPITCH];    // per-wave P tile, [row][key]

  const int tid    = threadIdx.x;
  const int wv     = tid >> 5;
  const int lane   = tid & 31;
  const int lanelo = lane & 15;
  const int half   = lane >> 4;

  const int qb = blockIdx.x * FQB;
  const int bh = blockIdx.y;
  const int b  = bh / Hq;
  const int h  = bh % Hq;
  const float scale = 0.125f;  // 1/sqrt(DH)

  // stage Q (pre-scaled), vectorized: float4 -> bf16x4
  #pragma unroll
  for (int it = 0; it < 8; ++it) {
    int idx = tid + it * 256;            // 0..2047
    int i = idx >> 4, g = idx & 15;
    float4 v = *(const float4*)&Qg[((size_t)(b * Sq + qb + i)) * Dq + h * DHq + 4 * g];
    store_bf4(&Qs[i][4 * g], v.x * scale, v.y * scale, v.z * scale, v.w * scale);
  }
  __syncthreads();

  bf16x16 aq[2];
  #pragma unroll
  for (int ks = 0; ks < 2; ++ks)
    aq[ks] = load_a_frag(&Qs[wv * 16 + lanelo][ks * 32], half);

  f32x8 o[4];
  #pragma unroll
  for (int j = 0; j < 4; ++j)
    #pragma unroll
    for (int e = 0; e < 8; ++e) o[j][e] = 0.f;
  float m_run[8], l_run[8];
  #pragma unroll
  for (int r = 0; r < 8; ++r) { m_run[r] = -1e30f; l_run[r] = 0.f; }

  const int kend = qb + FQB;  // causal: skip fully-masked key blocks
  for (int k0 = 0; k0 < kend; k0 += FKB) {
    __syncthreads();
    // K tile: row-major, vectorized
    #pragma unroll
    for (int it = 0; it < 4; ++it) {
      int idx = tid + it * 256;          // 0..1023
      int i = idx >> 4, g = idx & 15;
      float4 v = *(const float4*)&Kg[((size_t)(b * Sq + k0 + i)) * Dq + h * DHq + 4 * g];
      store_bf4(&Ks[i][4 * g], v.x, v.y, v.z, v.w);
    }
    // V tile: transposed (key-innermost), 4 strided loads -> one b64 store
    #pragma unroll
    for (int it = 0; it < 4; ++it) {
      int idx = tid + it * 256;          // 0..1023
      int dh = idx & 63, g = idx >> 6;   // g = key group of 4
      float v0 = Vg[((size_t)(b * Sq + k0 + 4 * g + 0)) * Dq + h * DHq + dh];
      float v1 = Vg[((size_t)(b * Sq + k0 + 4 * g + 1)) * Dq + h * DHq + dh];
      float v2 = Vg[((size_t)(b * Sq + k0 + 4 * g + 2)) * Dq + h * DHq + dh];
      float v3 = Vg[((size_t)(b * Sq + k0 + 4 * g + 3)) * Dq + h * DHq + dh];
      store_bf4(&VsT[dh][4 * g], v0, v1, v2, v3);
    }
    __syncthreads();

    // S = (Q*scale) @ K^T  (16x64 strip per wave)
    f32x8 sacc[4];
    #pragma unroll
    for (int j = 0; j < 4; ++j)
      #pragma unroll
      for (int e = 0; e < 8; ++e) sacc[j][e] = 0.f;
    #pragma unroll
    for (int ks = 0; ks < 2; ++ks) {
      bf16x16 bk[4];
      #pragma unroll
      for (int j = 0; j < 4; ++j)
        bk[j] = load_b_frag(&Ks[j * 16 + lanelo][ks * 32], half);
      #pragma unroll
      for (int j = 0; j < 4; ++j)
        sacc[j] = __builtin_amdgcn_wmma_f32_16x16x32_bf16(
            false, aq[ks], false, bk[j], (short)0, sacc[j], false, false);
    }

    // causal mask only on diagonal blocks (uniform per-wave predicate)
    if (k0 + FKB - 1 > qb + wv * 16) {
      #pragma unroll
      for (int r = 0; r < 8; ++r) {
        const int rowA = qb + wv * 16 + r + 8 * half;
        #pragma unroll
        for (int j = 0; j < 4; ++j) {
          int col = k0 + j * 16 + lanelo;
          if (col > rowA) sacc[j][r] = -1e30f;
        }
      }
    }

    // online softmax (rows live across the 16-lane half-wave)
    #pragma unroll
    for (int r = 0; r < 8; ++r) {
      float mx = m_run[r];
      #pragma unroll
      for (int j = 0; j < 4; ++j) mx = fmaxf(mx, sacc[j][r]);
      #pragma unroll
      for (int off = 8; off > 0; off >>= 1)
        mx = fmaxf(mx, __shfl_xor(mx, off, 16));
      float fac = __expf(m_run[r] - mx);
      float rs  = 0.f;
      #pragma unroll
      for (int j = 0; j < 4; ++j) {
        float pv = __expf(sacc[j][r] - mx);
        rs += pv;
        Ps[wv][r + 8 * half][j * 16 + lanelo] = (__bf16)pv;
      }
      #pragma unroll
      for (int off = 8; off > 0; off >>= 1) rs += __shfl_xor(rs, off, 16);
      m_run[r] = mx;
      l_run[r] = l_run[r] * fac + rs;
      #pragma unroll
      for (int j = 0; j < 4; ++j) o[j][r] *= fac;
    }

    // O += P @ V  (contraction over 64 keys = 2 WMMA K-steps)
    #pragma unroll
    for (int ks = 0; ks < 2; ++ks) {
      bf16x16 ap = load_a_frag(&Ps[wv][lanelo][ks * 32], half);
      bf16x16 bv[4];
      #pragma unroll
      for (int j = 0; j < 4; ++j)
        bv[j] = load_b_frag(&VsT[j * 16 + lanelo][ks * 32], half);
      #pragma unroll
      for (int j = 0; j < 4; ++j)
        o[j] = __builtin_amdgcn_wmma_f32_16x16x32_bf16(
            false, ap, false, bv[j], (short)0, o[j], false, false);
    }
  }

  float inv_l[8];
  #pragma unroll
  for (int r = 0; r < 8; ++r) inv_l[r] = __frcp_rn(l_run[r]);
  #pragma unroll
  for (int j = 0; j < 4; ++j) {
    int dh = j * 16 + lanelo;
    #pragma unroll
    for (int r = 0; r < 8; ++r) {
      int row = qb + wv * 16 + r + 8 * half;
      Ctx[((size_t)(b * Sq + row)) * Dq + h * DHq + dh] = o[j][r] * inv_l[r];
    }
  }
}

// ============================================================================
// LayerNorm over D=512, one row per 256-thread block (2 elems/thread, b64 IO)
// ============================================================================
__global__ __launch_bounds__(256) void ln512_kernel(
    const float* __restrict__ X, const float* __restrict__ g,
    const float* __restrict__ bta, float* __restrict__ Y)
{
  const size_t row = blockIdx.x;
  const float* xr = X + row * Dq;
  float*       yr = Y + row * Dq;
  const int t  = threadIdx.x;
  const int wv = t >> 5, ln = t & 31;

  float2 xv = *(const float2*)&xr[2 * t];
  float s = xv.x + xv.y;
  #pragma unroll
  for (int off = 16; off > 0; off >>= 1) s += __shfl_xor(s, off, 32);
  __shared__ float red[8];
  if (ln == 0) red[wv] = s;
  __syncthreads();
  float mean = (red[0] + red[1] + red[2] + red[3] +
                red[4] + red[5] + red[6] + red[7]) * (1.f / (float)Dq);

  float d0 = xv.x - mean, d1 = xv.y - mean;
  float s2 = d0 * d0 + d1 * d1;
  #pragma unroll
  for (int off = 16; off > 0; off >>= 1) s2 += __shfl_xor(s2, off, 32);
  __syncthreads();
  if (ln == 0) red[wv] = s2;
  __syncthreads();
  float var = (red[0] + red[1] + red[2] + red[3] +
               red[4] + red[5] + red[6] + red[7]) * (1.f / (float)Dq);
  float rstd = rsqrtf(var + EPSq);

  float2 gv = *(const float2*)&g[2 * t];
  float2 bv = *(const float2*)&bta[2 * t];
  float2 yv;
  yv.x = d0 * rstd * gv.x + bv.x;
  yv.y = d1 * rstd * gv.y + bv.y;
  *(float2*)&yr[2 * t] = yv;
}

// ============================================================================
extern "C" void kernel_launch(void* const* d_in, const int* in_sizes, int n_in,
                              void* d_out, int out_size, void* d_ws, size_t ws_size,
                              hipStream_t stream)
{
  (void)in_sizes; (void)n_in; (void)out_size; (void)ws_size;

  const float* x   = (const float*)d_in[0];
  // d_in[1] = mask: causal triu(1) -> handled analytically in flash kernel
  const float* wq  = (const float*)d_in[2];
  const float* bqv = (const float*)d_in[3];
  const float* wk  = (const float*)d_in[4];
  const float* bkv = (const float*)d_in[5];
  const float* wvm = (const float*)d_in[6];
  const float* bvv = (const float*)d_in[7];
  const float* wo  = (const float*)d_in[8];
  const float* bov = (const float*)d_in[9];
  const float* g1  = (const float*)d_in[10];
  const float* be1 = (const float*)d_in[11];
  const float* w1  = (const float*)d_in[12];
  const float* bb1 = (const float*)d_in[13];
  const float* w2  = (const float*)d_in[14];
  const float* bb2 = (const float*)d_in[15];
  const float* g2  = (const float*)d_in[16];
  const float* be2 = (const float*)d_in[17];

  const size_t BS = (size_t)Bq * Sq;  // 8192 rows
  float* ws  = (float*)d_ws;
  float* qb  = ws;                    // [BS, D]
  float* kb  = qb  + BS * Dq;
  float* vb  = kb  + BS * Dq;
  float* ctx = vb  + BS * Dq;
  float* z1  = ctx + BS * Dq;         // x + attn_out
  float* yb  = z1  + BS * Dq;         // LN1 output
  float* hb  = yb  + BS * Dq;         // [BS, F] ffn hidden
  float* z2  = hb  + BS * Fq;         // y + ffn_out

  dim3 blk(256);
  dim3 gP(Dq / GBN, (unsigned)(BS / GBM));   // 8 x 64 blocks
  gemm_bf16_wmma<<<gP, blk, 0, stream>>>(x,  wq,  bqv, nullptr, qb, (int)BS, Dq, Dq, 0);
  gemm_bf16_wmma<<<gP, blk, 0, stream>>>(x,  wk,  bkv, nullptr, kb, (int)BS, Dq, Dq, 0);
  gemm_bf16_wmma<<<gP, blk, 0, stream>>>(x,  wvm, bvv, nullptr, vb, (int)BS, Dq, Dq, 0);

  dim3 gA(Sq / FQB, Bq * Hq);                // 32 x 16 blocks
  flash_attn_wmma<<<gA, blk, 0, stream>>>(qb, kb, vb, ctx);

  gemm_bf16_wmma<<<gP, blk, 0, stream>>>(ctx, wo, bov, x, z1, (int)BS, Dq, Dq, 0);

  dim3 gL((unsigned)BS);
  ln512_kernel<<<gL, blk, 0, stream>>>(z1, g1, be1, yb);

  dim3 gF(Fq / GBN, (unsigned)(BS / GBM));   // 32 x 64 blocks
  gemm_bf16_wmma<<<gF, blk, 0, stream>>>(yb, w1, bb1, nullptr, hb, (int)BS, Fq, Dq, 1);
  gemm_bf16_wmma<<<gP, blk, 0, stream>>>(hb, w2, bb2, yb, z2, (int)BS, Dq, Fq, 0);

  ln512_kernel<<<gL, blk, 0, stream>>>(z2, g2, be2, (float*)d_out);
}
